// DeepSeekAttention_14439680049757
// MI455X (gfx1250) — matrix operand order repaired and checked by reading, not compiled, not run on hardware
//
#include <hip/hip_runtime.h>

// ---------------------------------------------------------------------------
// Types matching CDNA5 WMMA builtins (wave32, 16x16x32 bf16 -> f32)
// ---------------------------------------------------------------------------
typedef __attribute__((ext_vector_type(16))) __bf16 bf16x16;
typedef __attribute__((ext_vector_type(8)))  float  floatx8;
typedef __attribute__((ext_vector_type(4)))  int    v4i_t;

union FragAB {
    bf16x16 v;
    uint4   q[2];
    unsigned short us[16];
};

__device__ inline uint4 ldg4(const unsigned short* p) {
    return *reinterpret_cast<const uint4*>(p);
}

__device__ inline unsigned short f2bf(float f) {
    unsigned u = __float_as_uint(f);
    u += 0x7FFFu + ((u >> 16) & 1u);   // round-to-nearest-even
    return (unsigned short)(u >> 16);
}
__device__ inline float bf2f(unsigned short s) {
    return __uint_as_float(((unsigned)s) << 16);
}

__device__ inline floatx8 wmma_bf16(const FragAB& a, const FragAB& b, floatx8 c) {
    return __builtin_amdgcn_wmma_f32_16x16x32_bf16(
        false, a.v, false, b.v, (short)0, c, false, false);
}

// Async global->LDS copy support (gfx1250 GLOBAL_LOAD_ASYNC_TO_LDS_B128)
#if __has_builtin(__builtin_amdgcn_global_load_async_to_lds_b128)
#define HAS_ASYNC_LDS 1
typedef __attribute__((address_space(1))) v4i_t* g_v4i_ptr;   // prints as "__device__"
typedef __attribute__((address_space(3))) v4i_t* l_v4i_ptr;   // prints as "__shared__"
template <int ByteOff>
__device__ inline void async_cp16(const unsigned short* g, unsigned short* l) {
    __builtin_amdgcn_global_load_async_to_lds_b128(
        (g_v4i_ptr)(unsigned short*)g,   // drop const, cast to AS1 int4*
        (l_v4i_ptr)l,                    // cast to AS3 int4*
        ByteOff, 0);
}
#else
#define HAS_ASYNC_LDS 0
#endif

// ---------------------------------------------------------------------------
// Kernel 1: f32 -> bf16 convert, 8 elements / thread
// ---------------------------------------------------------------------------
__global__ __launch_bounds__(256) void f32_to_bf16_kernel(
    const float* __restrict__ in, unsigned short* __restrict__ out, int n8)
{
    int idx = blockIdx.x * 256 + threadIdx.x;
    if (idx >= n8) return;
    const float4* p = reinterpret_cast<const float4*>(in) + (size_t)idx * 2;
    float4 f0 = p[0], f1 = p[1];
    uint4 o;
    o.x = (unsigned)f2bf(f0.x) | ((unsigned)f2bf(f0.y) << 16);
    o.y = (unsigned)f2bf(f0.z) | ((unsigned)f2bf(f0.w) << 16);
    o.z = (unsigned)f2bf(f1.x) | ((unsigned)f2bf(f1.y) << 16);
    o.w = (unsigned)f2bf(f1.z) | ((unsigned)f2bf(f1.w) << 16);
    reinterpret_cast<uint4*>(out)[idx] = o;
}

// ---------------------------------------------------------------------------
// Kernel 2/5: C[M,N] = A[M,K] @ W[N,K]^T   (bf16 in, f32 accum)
// Each wave: 64x64 tile (4 M-frags x 4 N-frags = 16 WMMA / 32-wide k-step).
// ---------------------------------------------------------------------------
__global__ __launch_bounds__(256) void gemm_bf16_wmma(
    const unsigned short* __restrict__ A,
    const unsigned short* __restrict__ W,
    unsigned short* __restrict__ outB,   // bf16 output (or null)
    float* __restrict__ outF,            // f32 output (or null)
    int M, int N, int K)
{
    const int lane = threadIdx.x & 31;
    const int lm = lane & 15;
    const int lh = lane >> 4;
    const int widx = blockIdx.x * 8 + (threadIdx.x >> 5);
    const int tilesM = M >> 6;
    const int wm = widx % tilesM;
    const int wn = widx / tilesM;
    const int row0 = wm * 64;
    const int col0 = wn * 64;

    floatx8 zf = {0.f,0.f,0.f,0.f,0.f,0.f,0.f,0.f};
    floatx8 c[4][4];
    for (int i = 0; i < 4; ++i)
        for (int j = 0; j < 4; ++j) c[i][j] = zf;

    const unsigned short* aRow[4];
    const unsigned short* bRow[4];
    for (int i = 0; i < 4; ++i)
        aRow[i] = A + (size_t)(row0 + 16 * i + lm) * K;
    for (int j = 0; j < 4; ++j)
        bRow[j] = W + (size_t)(col0 + 16 * j + lm) * K;

    for (int k0 = 0; k0 < K; k0 += 32) {
        FragAB a[4], bq[4];
        // A-frag: lane=m, lanes 0-15 hold K {0..7,16..23}, lanes 16-31 {8..15,24..31}
        for (int i = 0; i < 4; ++i) {
            a[i].q[0] = ldg4(aRow[i] + k0 + lh * 8);
            a[i].q[1] = ldg4(aRow[i] + k0 + 16 + lh * 8);
        }
        // B-frag: lane%16 = n (row of W), same K order as A-frag
        for (int j = 0; j < 4; ++j) {
            bq[j].q[0] = ldg4(bRow[j] + k0 + lh * 8);
            bq[j].q[1] = ldg4(bRow[j] + k0 + 16 + lh * 8);
        }
        for (int i = 0; i < 4; ++i)
            for (int j = 0; j < 4; ++j)
                c[i][j] = wmma_bf16(a[i], bq[j], c[i][j]);
    }

    for (int i = 0; i < 4; ++i) {
        for (int j = 0; j < 4; ++j) {
            for (int r = 0; r < 8; ++r) {
                int row = row0 + 16 * i + r + 8 * lh;
                int col = col0 + 16 * j + lm;
                size_t off = (size_t)row * N + col;
                if (outF) outF[off] = c[i][j][r];
                else      outB[off] = f2bf(c[i][j][r]);
            }
        }
    }
}

// ---------------------------------------------------------------------------
// Kernel 3: RoPE in place on Q and K (layout [B,S,nh,hd], bf16)
// ---------------------------------------------------------------------------
__global__ __launch_bounds__(256) void rope_kernel(
    unsigned short* __restrict__ Q, unsigned short* __restrict__ K)
{
    int idx = blockIdx.x * 256 + threadIdx.x;     // B*S*16*64 = 4M
    int i = idx & 63;
    int rest = idx >> 6;                          // (b*S+s)*16 + h
    int s = (rest >> 4) & 2047;
    float ang = (float)s * __expf(-0.14391156831f * (float)i); // 10000^(-i/64)
    float sn, cs;
    __sincosf(ang, &sn, &cs);
    size_t base = (size_t)rest * 128 + i;
    float q0 = bf2f(Q[base]), q1 = bf2f(Q[base + 64]);
    Q[base]      = f2bf(q0 * cs - q1 * sn);
    Q[base + 64] = f2bf(q1 * cs + q0 * sn);
    float k0 = bf2f(K[base]), k1 = bf2f(K[base + 64]);
    K[base]      = f2bf(k0 * cs - k1 * sn);
    K[base + 64] = f2bf(k1 * cs + k0 * sn);
}

// ---------------------------------------------------------------------------
// Kernel 4: causal flash attention, bf16 WMMA, f32 online softmax.
// Block = 128 threads (4 waves); each wave owns 16 query rows.
// K tile staged via async global->LDS (overlaps with manual V transpose).
// ---------------------------------------------------------------------------
#define SQ 2048
#define KLDS_STRIDE 144   // 128 + 16 pad (ushorts), 16B-multiple
#define VLDS_STRIDE 48    // 32 + 16 pad
#define PLDS_STRIDE 48

__global__ __launch_bounds__(128) void attn_wmma(
    const unsigned short* __restrict__ Qb,
    const unsigned short* __restrict__ Kb,
    const unsigned short* __restrict__ Vb,
    unsigned short* __restrict__ Ob)
{
    __shared__ unsigned short Klds[32 * KLDS_STRIDE];     // [key][d]
    __shared__ unsigned short Vlds[128 * VLDS_STRIDE];    // [d][key] (transposed)
    __shared__ unsigned short Plds[4][16 * PLDS_STRIDE];  // per-wave P staging

    const int tid = threadIdx.x;
    const int lane = tid & 31;
    const int wv = tid >> 5;
    const int lm = lane & 15;
    const int lh = lane >> 4;

    const int qblk = blockIdx.x & 31;
    const int h    = (blockIdx.x >> 5) & 15;
    const int b    = blockIdx.x >> 9;

    const int qbase = qblk * 64 + wv * 16;
    const size_t batchOff = (size_t)b * SQ * 2048;

    // Preload Q fragments (16 rows x 128 head-dim = 4 A-frags), resident
    FragAB aq[4];
    {
        const unsigned short* qrow = Qb + batchOff + (size_t)(qbase + lm) * 2048 + h * 128;
        for (int f = 0; f < 4; ++f) {
            aq[f].q[0] = ldg4(qrow + 32 * f + lh * 8);
            aq[f].q[1] = ldg4(qrow + 32 * f + 16 + lh * 8);
        }
    }

    floatx8 zf = {0.f,0.f,0.f,0.f,0.f,0.f,0.f,0.f};
    floatx8 o[8];
    for (int cdx = 0; cdx < 8; ++cdx) o[cdx] = zf;
    float mrow[8], lrow[8];
    for (int r = 0; r < 8; ++r) { mrow[r] = -1e30f; lrow[r] = 0.f; }

    const int nkt = 2 * qblk + 2;      // key tiles of 32 up to block diagonal
    const float scale = 0.0883883476483184f;   // 1/sqrt(128)

    // Per-thread K-tile staging addresses (64B chunk per thread)
    const int kk = tid >> 2;          // key row 0..31
    const int quarter = tid & 3;      // 0..3 (32 elems each)
    unsigned short* kdst = &Klds[kk * KLDS_STRIDE + quarter * 32];

    for (int kt = 0; kt < nkt; ++kt) {
        __syncthreads();
        // ---- stage K tile (32 keys x 128) row-major ----
        {
            const unsigned short* krow =
                Kb + batchOff + (size_t)(kt * 32 + kk) * 2048 + h * 128 + quarter * 32;
#if HAS_ASYNC_LDS
            // offset immediate is added to BOTH global and LDS addresses
            async_cp16<0>(krow, kdst);
            async_cp16<16>(krow, kdst);
            async_cp16<32>(krow, kdst);
            async_cp16<48>(krow, kdst);
#else
            uint4 t0 = ldg4(krow);
            uint4 t1 = ldg4(krow + 8);
            uint4 t2 = ldg4(krow + 16);
            uint4 t3 = ldg4(krow + 24);
            *reinterpret_cast<uint4*>(kdst)      = t0;
            *reinterpret_cast<uint4*>(kdst + 8)  = t1;
            *reinterpret_cast<uint4*>(kdst + 16) = t2;
            *reinterpret_cast<uint4*>(kdst + 24) = t3;
#endif
        }
        // ---- stage V tile transposed [d][key] via packed-dword transpose ----
        for (int task = tid; task < 256; task += 128) {
            int p  = task & 15;      // key pair
            int cd = task >> 4;      // d chunk of 8
            int k0 = 2 * p, d0 = cd * 8;
            const unsigned short* v0 =
                Vb + batchOff + (size_t)(kt * 32 + k0) * 2048 + h * 128 + d0;
            uint4 r0 = ldg4(v0);
            uint4 r1 = ldg4(v0 + 2048);
            const unsigned short* s0 = reinterpret_cast<const unsigned short*>(&r0);
            const unsigned short* s1 = reinterpret_cast<const unsigned short*>(&r1);
            for (int e = 0; e < 8; ++e) {
                unsigned dw = (unsigned)s0[e] | ((unsigned)s1[e] << 16);
                *reinterpret_cast<unsigned*>(&Vlds[(d0 + e) * VLDS_STRIDE + k0]) = dw;
            }
        }
#if HAS_ASYNC_LDS
        asm volatile("s_wait_asynccnt 0" ::: "memory");
#endif
        __syncthreads();

        // ---- scores: S = Q(16x128) @ K_tile^T -> two 16x16 C frags ----
        floatx8 s0 = zf, s1 = zf;
        for (int ks = 0; ks < 4; ++ks) {
            FragAB b0, b1;
            const unsigned short* kb0 = &Klds[lm * KLDS_STRIDE + ks * 32 + lh * 8];
            const unsigned short* kb1 = &Klds[(16 + lm) * KLDS_STRIDE + ks * 32 + lh * 8];
            b0.q[0] = *reinterpret_cast<const uint4*>(kb0);
            b0.q[1] = *reinterpret_cast<const uint4*>(kb0 + 16);
            b1.q[0] = *reinterpret_cast<const uint4*>(kb1);
            b1.q[1] = *reinterpret_cast<const uint4*>(kb1 + 16);
            s0 = wmma_bf16(aq[ks], b0, s0);
            s1 = wmma_bf16(aq[ks], b1, s1);
        }

        // ---- online softmax (row = r + 8*lh, col = lane%16 within half) ----
        const int ktbase = kt * 32;
        const bool needMask = (ktbase + 31) > qbase;
        float pr0[8], pr1[8];
        for (int r = 0; r < 8; ++r) {
            float va = s0[r] * scale;
            float vb = s1[r] * scale;
            bool m0 = false, m1 = false;
            if (needMask) {
                int qrow = qbase + r + 8 * lh;
                int key0 = ktbase + lm;
                m0 = key0 > qrow;
                m1 = (key0 + 16) > qrow;
            }
            float xa = m0 ? -1e30f : va;
            float xb = m1 ? -1e30f : vb;
            float vmax = fmaxf(xa, xb);
            for (int off = 1; off < 16; off <<= 1)
                vmax = fmaxf(vmax, __shfl_xor(vmax, off, 32));
            float newm = fmaxf(mrow[r], vmax);
            float fac = __expf(mrow[r] - newm);
            mrow[r] = newm;
            float p0 = m0 ? 0.f : __expf(va - newm);
            float p1 = m1 ? 0.f : __expf(vb - newm);
            float rs = p0 + p1;
            for (int off = 1; off < 16; off <<= 1)
                rs += __shfl_xor(rs, off, 32);
            lrow[r] = lrow[r] * fac + rs;
            for (int cdx = 0; cdx < 8; ++cdx) o[cdx][r] *= fac;
            pr0[r] = p0; pr1[r] = p1;
        }

        // ---- P: C-layout -> A-layout via per-wave LDS (LDS in-order per wave)
        unsigned short* pw = Plds[wv];
        asm volatile("" ::: "memory");
        for (int r = 0; r < 8; ++r) {
            int m = r + 8 * lh;
            pw[m * PLDS_STRIDE + lm]      = f2bf(pr0[r]);
            pw[m * PLDS_STRIDE + 16 + lm] = f2bf(pr1[r]);
        }
        asm volatile("" ::: "memory");
        FragAB pa;
        {
            const unsigned short* pr = &pw[lm * PLDS_STRIDE + lh * 8];
            pa.q[0] = *reinterpret_cast<const uint4*>(pr);
            pa.q[1] = *reinterpret_cast<const uint4*>(pr + 16);
        }

        // ---- O += P(16x32) @ V(32x128), 8 d-chunks ----
        for (int cdx = 0; cdx < 8; ++cdx) {
            FragAB vbf;
            const unsigned short* vr = &Vlds[(cdx * 16 + lm) * VLDS_STRIDE + lh * 8];
            vbf.q[0] = *reinterpret_cast<const uint4*>(vr);
            vbf.q[1] = *reinterpret_cast<const uint4*>(vr + 16);
            o[cdx] = wmma_bf16(pa, vbf, o[cdx]);
        }
    }

    // ---- normalize and store (attention output, bf16, [M, 2048] layout) ----
    unsigned short* orow = Ob + batchOff + (size_t)qbase * 2048 + h * 128;
    for (int r = 0; r < 8; ++r) {
        float inv = 1.0f / lrow[r];
        int m = r + 8 * lh;
        for (int cdx = 0; cdx < 8; ++cdx) {
            int d = cdx * 16 + lm;
            orow[(size_t)m * 2048 + d] = f2bf(o[cdx][r] * inv);
        }
    }
}

// ---------------------------------------------------------------------------
// Launch
// ---------------------------------------------------------------------------
extern "C" void kernel_launch(void* const* d_in, const int* in_sizes, int n_in,
                              void* d_out, int out_size, void* d_ws, size_t ws_size,
                              hipStream_t stream)
{
    const float* x  = (const float*)d_in[0];
    const float* Wq = (const float*)d_in[1];
    const float* Wk = (const float*)d_in[2];
    const float* Wv = (const float*)d_in[3];
    const float* Wo = (const float*)d_in[4];
    float* out = (float*)d_out;

    const size_t MB = 1024 * 1024;
    char* ws = (char*)d_ws;
    unsigned short* xb  = (unsigned short*)(ws);             // 16 MB  (4096x2048)
    unsigned short* wqb = (unsigned short*)(ws + 16 * MB);   //  8 MB
    unsigned short* wkb = (unsigned short*)(ws + 24 * MB);
    unsigned short* wvb = (unsigned short*)(ws + 32 * MB);
    unsigned short* wob = (unsigned short*)(ws + 40 * MB);
    unsigned short* qb  = (unsigned short*)(ws + 48 * MB);   // 16 MB each
    unsigned short* kb  = (unsigned short*)(ws + 64 * MB);
    unsigned short* vb  = (unsigned short*)(ws + 80 * MB);
    unsigned short* ob  = (unsigned short*)(ws + 96 * MB);

    // 1) convert to bf16
    f32_to_bf16_kernel<<<4096, 256, 0, stream>>>(x,  xb, 1048576); // 8M elems
    f32_to_bf16_kernel<<<2048, 256, 0, stream>>>(Wq, wqb, 524288); // 4M elems
    f32_to_bf16_kernel<<<2048, 256, 0, stream>>>(Wk, wkb, 524288);
    f32_to_bf16_kernel<<<2048, 256, 0, stream>>>(Wv, wvb, 524288);
    f32_to_bf16_kernel<<<2048, 256, 0, stream>>>(Wo, wob, 524288);

    // 2) Q/K/V projections (x @ W^T), output [4096, 2048] == [B,S,nh,hd]
    gemm_bf16_wmma<<<256, 256, 0, stream>>>(xb, wqb, qb, nullptr, 4096, 2048, 2048);
    gemm_bf16_wmma<<<256, 256, 0, stream>>>(xb, wkb, kb, nullptr, 4096, 2048, 2048);
    gemm_bf16_wmma<<<256, 256, 0, stream>>>(xb, wvb, vb, nullptr, 4096, 2048, 2048);

    // 3) RoPE on Q,K in place  (B*S*16*64 = 4M threads)
    rope_kernel<<<16384, 256, 0, stream>>>(qb, kb);

    // 4) causal flash attention  (B * nh * S/64 = 1024 blocks)
    attn_wmma<<<1024, 128, 0, stream>>>(qb, kb, vb, ob);

    // 5) output projection, f32 store
    gemm_bf16_wmma<<<256, 256, 0, stream>>>(ob, wob, nullptr, out, 4096, 2048, 2048);
}
